// GQA_89610197664136
// MI455X (gfx1250) — compile-verified, matrix operand order checked by reading, not run on hardware
//
#include <hip/hip_runtime.h>
#include <hip/hip_bf16.h>

typedef __bf16 bf16_t;
typedef bf16_t v8bf  __attribute__((ext_vector_type(8)));
typedef bf16_t v16bf __attribute__((ext_vector_type(16)));
typedef float  v8f   __attribute__((ext_vector_type(8)));

static __device__ __forceinline__ v16bf cat16(v8bf a, v8bf b) {
  return __builtin_shufflevector(a, b, 0,1,2,3,4,5,6,7,8,9,10,11,12,13,14,15);
}
#define WMMA_BF16(a, b, c) \
  __builtin_amdgcn_wmma_f32_16x16x32_bf16(false, (a), false, (b), (short)0, (c), false, false)

// CDNA5 async global->LDS copy (per-lane 16B), tracked by ASYNCcnt.
static __device__ __forceinline__ void async_b128(unsigned lds_off, unsigned g_off,
                                                  unsigned long long gbase) {
  asm volatile("global_load_async_to_lds_b128 %0, %1, %2"
               :: "v"(lds_off), "v"(g_off), "s"(gbase) : "memory");
}
static __device__ __forceinline__ void wait_async0() {
  asm volatile("s_wait_asynccnt 0" ::: "memory");
}
static __device__ __forceinline__ unsigned lds_addr(const void* p) {
  return (unsigned)(uintptr_t)p;   // low 32 bits of flat shared address = LDS offset
}

// Problem constants
constexpr int Bc   = 2;
constexpr int Sc   = 2048;
constexpr int DIMc = 1024;
constexpr int NH   = 16;
constexpr int NKV  = 4;
constexpr int HD   = 64;
constexpr int KVD  = NKV * HD;           // 256
constexpr int NQKV = DIMc + 2 * KVD;     // 1536
constexpr int MR   = Bc * Sc;            // 4096 rows

// ---------------------------------------------------------------- converts
__global__ void cvt_f32_bf16(const float* __restrict__ in, bf16_t* __restrict__ out, int n) {
  for (int i = blockIdx.x * blockDim.x + threadIdx.x; i < n; i += gridDim.x * blockDim.x)
    out[i] = (bf16_t)in[i];
}
// in[K][N] (f32) -> out[N][K] (bf16)
__global__ void cvt_transpose(const float* __restrict__ in, bf16_t* __restrict__ out,
                              int K, int N) {
  int total = K * N;
  for (int i = blockIdx.x * blockDim.x + threadIdx.x; i < total; i += gridDim.x * blockDim.x) {
    int k = i / N, n = i - k * N;
    out[(size_t)n * K + k] = (bf16_t)in[i];
  }
}

// ---------------------------------------------------------------- GEMM staging
// A tile: 128 rows x 32 k from Ag (row-major, ld=1024). B tile: 128 n-rows x 32 k
// from Bg = W^T (row-major, ld=1024). Both land in [row][k] LDS layout, pitch 40.
static __device__ __forceinline__ void stage_ab_async(
    unsigned ldsA, unsigned ldsB, unsigned long long ag, unsigned long long bg,
    int rowBase, int colBase, int kk, int tid) {
#pragma unroll
  for (int i = 0; i < 2; ++i) {
    int c  = tid * 2 + i;          // 512 16B-chunks
    int r  = c >> 2;
    int k0 = (c & 3) * 8;
    async_b128(ldsA + (unsigned)(r * 40 + k0) * 2,
               (unsigned)(((rowBase + r) * DIMc + kk * 32 + k0) * 2), ag);
    async_b128(ldsB + (unsigned)(r * 40 + k0) * 2,
               (unsigned)(((colBase + r) * DIMc + kk * 32 + k0) * 2), bg);
  }
}

// ---------------------------------------------------------------- QKV GEMM
// C[4096 x 1536] = Xb * WqkvT^T + bias, scattered to Q / K / V^T (bf16)
__global__ __launch_bounds__(256) void qkv_gemm(
    const bf16_t* __restrict__ Xb, const bf16_t* __restrict__ WT,
    const float* __restrict__ bias,
    bf16_t* __restrict__ Qb, bf16_t* __restrict__ Kb, bf16_t* __restrict__ VTb) {
  __shared__ bf16_t Alds[2][128 * 40];
  __shared__ bf16_t Blds[2][128 * 40];
  const int tid = threadIdx.x;
  const int lane = tid & 31, w = tid >> 5;
  const int m = lane & 15, hi = lane >> 4;
  const int wm = w & 3, wn = w >> 2;        // wave tile 32 x 64 in a 128 x 128 block tile
  const int rowBase = blockIdx.x * 128;
  const int colBase = blockIdx.y * 128;

  const unsigned aB = lds_addr(&Alds[0][0]);
  const unsigned bB = lds_addr(&Blds[0][0]);
  const unsigned long long ag = (unsigned long long)(uintptr_t)Xb;
  const unsigned long long bg = (unsigned long long)(uintptr_t)WT;

  stage_ab_async(aB, bB, ag, bg, rowBase, colBase, 0, tid);
  v8f c[2][4] = {};
  for (int kk = 0; kk < DIMc / 32; ++kk) {
    wait_async0();
    __syncthreads();
    if (kk + 1 < DIMc / 32) {
      unsigned boff = (unsigned)(((kk + 1) & 1) * 128 * 40 * 2);
      stage_ab_async(aB + boff, bB + boff, ag, bg, rowBase, colBase, kk + 1, tid);
    }
    const bf16_t* Ab = &Alds[kk & 1][0];
    const bf16_t* Bb = &Blds[kk & 1][0];
    v16bf a[2], b[4];
#pragma unroll
    for (int mi = 0; mi < 2; ++mi) {
      const bf16_t* p = Ab + (wm * 32 + mi * 16 + m) * 40 + hi * 8;
      a[mi] = cat16(*(const v8bf*)p, *(const v8bf*)(p + 16));
    }
#pragma unroll
    for (int ni = 0; ni < 4; ++ni) {
      const bf16_t* p = Bb + (wn * 64 + ni * 16 + m) * 40 + hi * 8;
      b[ni] = cat16(*(const v8bf*)p, *(const v8bf*)(p + 16));
    }
#pragma unroll
    for (int mi = 0; mi < 2; ++mi)
#pragma unroll
      for (int ni = 0; ni < 4; ++ni) c[mi][ni] = WMMA_BF16(a[mi], b[ni], c[mi][ni]);
  }
  // epilogue: bias + scatter (V is written transposed: [b][kv][d][s])
#pragma unroll
  for (int mi = 0; mi < 2; ++mi)
#pragma unroll
    for (int ni = 0; ni < 4; ++ni)
#pragma unroll
      for (int r = 0; r < 8; ++r) {
        int R  = rowBase + wm * 32 + mi * 16 + r + 8 * hi;
        int cx = colBase + wn * 64 + ni * 16 + m;
        float v = c[mi][ni][r] + bias[cx];
        int bb = R >> 11, s = R & 2047;
        bf16_t bv = (bf16_t)v;
        if (cx < DIMc) {
          int h = cx >> 6, d = cx & 63;
          Qb[(((size_t)bb * NH + h) * Sc + s) * HD + d] = bv;
        } else if (cx < DIMc + KVD) {
          int c2 = cx - DIMc, kv = c2 >> 6, d = c2 & 63;
          Kb[(((size_t)bb * NKV + kv) * Sc + s) * HD + d] = bv;
        } else {
          int c2 = cx - DIMc - KVD, kv = c2 >> 6, d = c2 & 63;
          VTb[(((size_t)bb * NKV + kv) * HD + d) * Sc + s] = bv;
        }
      }
}

// ---------------------------------------------------------------- flash attention
// block = (b, head, 64 q-rows); 4 waves x 16 rows; 32-key tiles, async double-buffered
__global__ __launch_bounds__(128) void attn_kernel(
    const bf16_t* __restrict__ Qb, const bf16_t* __restrict__ Kb,
    const bf16_t* __restrict__ VTb, bf16_t* __restrict__ Yb) {
  __shared__ bf16_t Klds[2][32 * 72];      // K[t][d] row-major (padded)
  __shared__ bf16_t Vlds[2][64 * 40];      // V^T[d][t]
  __shared__ bf16_t Plds[4 * 16 * 40];     // per-wave P[m][t]
  const int tid = threadIdx.x, lane = tid & 31, w = tid >> 5;
  const int m = lane & 15, hi = lane >> 4;
  const int qb = blockIdx.x & 31;
  const int h  = (blockIdx.x >> 5) & 15;
  const int b  = blockIdx.x >> 9;
  const int kv = h >> 2;
  const int qrow0 = qb * 64 + w * 16;

  const bf16_t* qp = Qb + (((size_t)b * NH + h) * Sc + qrow0 + m) * HD;
  v16bf qa[2];
#pragma unroll
  for (int kk = 0; kk < 2; ++kk)
    qa[kk] = cat16(*(const v8bf*)(qp + kk * 32 + hi * 8),
                   *(const v8bf*)(qp + kk * 32 + 16 + hi * 8));

  float mstat[8], lstat[8];
#pragma unroll
  for (int r = 0; r < 8; ++r) { mstat[r] = -1e30f; lstat[r] = 0.f; }
  v8f o[4] = {};

  const unsigned kB = lds_addr(&Klds[0][0]);
  const unsigned vB = lds_addr(&Vlds[0][0]);
  const unsigned long long kg = (unsigned long long)(uintptr_t)(Kb + ((size_t)b * NKV + kv) * Sc * HD);
  const unsigned long long vg = (unsigned long long)(uintptr_t)(VTb + ((size_t)b * NKV + kv) * (size_t)HD * Sc);

  const int nkt = qb * 2 + 2;   // causal: keys <= qb*64+63

  // issue tile 0
#pragma unroll
  for (int i = 0; i < 2; ++i) {
    int c = tid * 2 + i;
    int t = c >> 3, d0 = (c & 7) * 8;
    async_b128(kB + (unsigned)(t * 72 + d0) * 2, (unsigned)((t * HD + d0) * 2), kg);
    int d = c >> 2, t0 = (c & 3) * 8;
    async_b128(vB + (unsigned)(d * 40 + t0) * 2, (unsigned)((d * Sc + t0) * 2), vg);
  }

  for (int kt = 0; kt < nkt; ++kt) {
    const int kb = kt * 32;
    wait_async0();
    __syncthreads();
    if (kt + 1 < nkt) {  // prefetch next key tile into the other buffer
      int kb2 = kb + 32;
      unsigned kD = kB + (unsigned)(((kt + 1) & 1) * 32 * 72 * 2);
      unsigned vD = vB + (unsigned)(((kt + 1) & 1) * 64 * 40 * 2);
#pragma unroll
      for (int i = 0; i < 2; ++i) {
        int c = tid * 2 + i;
        int t = c >> 3, d0 = (c & 7) * 8;
        async_b128(kD + (unsigned)(t * 72 + d0) * 2, (unsigned)(((kb2 + t) * HD + d0) * 2), kg);
        int d = c >> 2, t0 = (c & 3) * 8;
        async_b128(vD + (unsigned)(d * 40 + t0) * 2, (unsigned)((d * Sc + kb2 + t0) * 2), vg);
      }
    }
    const bf16_t* Kt = &Klds[kt & 1][0];
    const bf16_t* Vt = &Vlds[kt & 1][0];

    // S = Q * K^T  (two 16-key column tiles)
    v8f s0 = {}, s1 = {};
#pragma unroll
    for (int kk = 0; kk < 2; ++kk) {
      const bf16_t* p0 = Kt + (0 * 16 + m) * 72 + kk * 32 + hi * 8;
      const bf16_t* p1 = Kt + (1 * 16 + m) * 72 + kk * 32 + hi * 8;
      s0 = WMMA_BF16(qa[kk], cat16(*(const v8bf*)p0, *(const v8bf*)(p0 + 16)), s0);
      s1 = WMMA_BF16(qa[kk], cat16(*(const v8bf*)p1, *(const v8bf*)(p1 + 16)), s1);
    }

    // online softmax update
#pragma unroll
    for (int r = 0; r < 8; ++r) {
      int qg = qrow0 + r + 8 * hi;
      float v0 = s0[r] * 0.125f;
      float v1 = s1[r] * 0.125f;
      if (kb + m > qg)      v0 = -1e30f;
      if (kb + 16 + m > qg) v1 = -1e30f;
      float mx = fmaxf(v0, v1);
      mx = fmaxf(mx, __shfl_xor(mx, 1, 32));
      mx = fmaxf(mx, __shfl_xor(mx, 2, 32));
      mx = fmaxf(mx, __shfl_xor(mx, 4, 32));
      mx = fmaxf(mx, __shfl_xor(mx, 8, 32));
      float mnew  = fmaxf(mstat[r], mx);
      float alpha = __expf(mstat[r] - mnew);
      float p0 = __expf(v0 - mnew);
      float p1 = __expf(v1 - mnew);
      float ps = p0 + p1;
      ps += __shfl_xor(ps, 1, 32);
      ps += __shfl_xor(ps, 2, 32);
      ps += __shfl_xor(ps, 4, 32);
      ps += __shfl_xor(ps, 8, 32);
      lstat[r] = lstat[r] * alpha + ps;
      mstat[r] = mnew;
      Plds[(w * 16 + r + 8 * hi) * 40 + m]      = (bf16_t)p0;
      Plds[(w * 16 + r + 8 * hi) * 40 + 16 + m] = (bf16_t)p1;
      o[0][r] *= alpha; o[1][r] *= alpha; o[2][r] *= alpha; o[3][r] *= alpha;
    }
    asm volatile("s_wait_dscnt 0" ::: "memory");

    // re-fragment P as WMMA A operand, then O += P * V
    const bf16_t* pp = Plds + (w * 16 + m) * 40 + hi * 8;
    v16bf pa = cat16(*(const v8bf*)pp, *(const v8bf*)(pp + 16));
#pragma unroll
    for (int nn = 0; nn < 4; ++nn) {
      const bf16_t* vp = Vt + (nn * 16 + m) * 40 + hi * 8;
      o[nn] = WMMA_BF16(pa, cat16(*(const v8bf*)vp, *(const v8bf*)(vp + 16)), o[nn]);
    }
  }

  // normalize and store Y (row-major [B*S][1024], col = h*64+d)
#pragma unroll
  for (int nn = 0; nn < 4; ++nn)
#pragma unroll
    for (int r = 0; r < 8; ++r) {
      int row = qrow0 + r + 8 * hi;
      int d   = nn * 16 + m;
      float v = o[nn][r] / lstat[r];
      Yb[((size_t)b * Sc + row) * DIMc + h * HD + d] = (bf16_t)v;
    }
}

// ---------------------------------------------------------------- out GEMM
// out[4096 x 1024] = Yb * WoutT^T + bias (f32 out)
__global__ __launch_bounds__(256) void out_gemm(
    const bf16_t* __restrict__ Yb, const bf16_t* __restrict__ WT,
    const float* __restrict__ bias, float* __restrict__ out) {
  __shared__ bf16_t Alds[2][128 * 40];
  __shared__ bf16_t Blds[2][128 * 40];
  const int tid = threadIdx.x;
  const int lane = tid & 31, w = tid >> 5;
  const int m = lane & 15, hi = lane >> 4;
  const int wm = w & 3, wn = w >> 2;
  const int rowBase = blockIdx.x * 128;
  const int colBase = blockIdx.y * 128;

  const unsigned aB = lds_addr(&Alds[0][0]);
  const unsigned bB = lds_addr(&Blds[0][0]);
  const unsigned long long ag = (unsigned long long)(uintptr_t)Yb;
  const unsigned long long bg = (unsigned long long)(uintptr_t)WT;

  stage_ab_async(aB, bB, ag, bg, rowBase, colBase, 0, tid);
  v8f c[2][4] = {};
  for (int kk = 0; kk < DIMc / 32; ++kk) {
    wait_async0();
    __syncthreads();
    if (kk + 1 < DIMc / 32) {
      unsigned boff = (unsigned)(((kk + 1) & 1) * 128 * 40 * 2);
      stage_ab_async(aB + boff, bB + boff, ag, bg, rowBase, colBase, kk + 1, tid);
    }
    const bf16_t* Ab = &Alds[kk & 1][0];
    const bf16_t* Bb = &Blds[kk & 1][0];
    v16bf a[2], b[4];
#pragma unroll
    for (int mi = 0; mi < 2; ++mi) {
      const bf16_t* p = Ab + (wm * 32 + mi * 16 + m) * 40 + hi * 8;
      a[mi] = cat16(*(const v8bf*)p, *(const v8bf*)(p + 16));
    }
#pragma unroll
    for (int ni = 0; ni < 4; ++ni) {
      const bf16_t* p = Bb + (wn * 64 + ni * 16 + m) * 40 + hi * 8;
      b[ni] = cat16(*(const v8bf*)p, *(const v8bf*)(p + 16));
    }
#pragma unroll
    for (int mi = 0; mi < 2; ++mi)
#pragma unroll
      for (int ni = 0; ni < 4; ++ni) c[mi][ni] = WMMA_BF16(a[mi], b[ni], c[mi][ni]);
  }
#pragma unroll
  for (int mi = 0; mi < 2; ++mi)
#pragma unroll
    for (int ni = 0; ni < 4; ++ni)
#pragma unroll
      for (int r = 0; r < 8; ++r) {
        int R  = rowBase + wm * 32 + mi * 16 + r + 8 * hi;
        int cx = colBase + wn * 64 + ni * 16 + m;
        out[(size_t)R * DIMc + cx] = c[mi][ni][r] + bias[cx];
      }
}

// ---------------------------------------------------------------- launch
extern "C" void kernel_launch(void* const* d_in, const int* in_sizes, int n_in,
                              void* d_out, int out_size, void* d_ws, size_t ws_size,
                              hipStream_t stream) {
  const float* x     = (const float*)d_in[0];
  const float* W_qkv = (const float*)d_in[1];
  const float* b_qkv = (const float*)d_in[2];
  const float* W_out = (const float*)d_in[3];
  const float* b_out = (const float*)d_in[4];
  float* out = (float*)d_out;

  char* ws = (char*)d_ws;
  // workspace layout (bytes)
  bf16_t* Xb  = (bf16_t*)(ws + 0);          // 4096*1024*2   =  8388608
  bf16_t* WqT = (bf16_t*)(ws + 8388608);    // 1536*1024*2   =  3145728  (W_qkv^T)
  bf16_t* WoT = (bf16_t*)(ws + 11534336);   // 1024*1024*2   =  2097152  (W_out^T)
  bf16_t* Qb  = (bf16_t*)(ws + 13631488);   // 4096*1024*2   =  8388608
  bf16_t* Kb  = (bf16_t*)(ws + 22020096);   // 2*4*2048*64*2 =  2097152
  bf16_t* VTb = (bf16_t*)(ws + 24117248);   // 2097152  (V^T: [b][kv][d][s])
  bf16_t* Yb  = (bf16_t*)(ws + 26214400);   // 8388608   (total 34603008 bytes)

  cvt_f32_bf16 <<<2048, 256, 0, stream>>>(x, Xb, MR * DIMc);
  cvt_transpose<<<2048, 256, 0, stream>>>(W_qkv, WqT, DIMc, NQKV);
  cvt_transpose<<<2048, 256, 0, stream>>>(W_out, WoT, DIMc, DIMc);

  qkv_gemm<<<dim3(MR / 128, NQKV / 128), 256, 0, stream>>>(Xb, WqT, b_qkv, Qb, Kb, VTb);
  attn_kernel<<<Bc * NH * (Sc / 64), 128, 0, stream>>>(Qb, Kb, VTb, Yb);
  out_gemm<<<dim3(MR / 128, DIMc / 128), 256, 0, stream>>>(Yb, WoT, b_out, out);
}
